// _netlocalD_63342177681511
// MI455X (gfx1250) — matrix operand order repaired and checked by reading, not compiled, not run on hardware
//
#include <hip/hip_runtime.h>
#include <cstdint>

// ---------------------------------------------------------------------------
// DGCNN-like forward for MI455X (gfx1250, wave32, WMMA).
//  - edge-conv GEMMs  (59 GF) -> v_wmma_f32_16x16x32_f16 (f16 weights preconv)
//  - kNN distance GEMM (17 GF) -> v_wmma_f32_16x16x4_f32, tiled to workspace;
//    top-k selection runs in a separate register-domain (no spills around WMMA)
// ---------------------------------------------------------------------------

typedef __attribute__((ext_vector_type(16))) _Float16 v16h;
typedef __attribute__((ext_vector_type(8)))  _Float16 v8h;
typedef __attribute__((ext_vector_type(8)))  float    v8f;
typedef __attribute__((ext_vector_type(2)))  float    v2f;

static constexpr int Bn    = 8;     // batch
static constexpr int Nn    = 2048;  // points
static constexpr int KNN   = 20;    // neighbors
static constexpr int CHUNK = 512;   // kNN candidate chunk (dist buffer = 33.5 MB)

// ---------------------------------------------------------------------------
__global__ void zero_f32(float* __restrict__ p, int n) {
  int i = blockIdx.x * blockDim.x + threadIdx.x;
  if (i < n) p[i] = 0.0f;
}

__global__ void cvt_w_f16(const float* __restrict__ W, _Float16* __restrict__ Wh, int n) {
  int i = blockIdx.x * blockDim.x + threadIdx.x;
  if (i < n) Wh[i] = (_Float16)W[i];
}

// ---------------------------------------------------------------------------
// Per-point squared norms (used by the WMMA kNN)
// ---------------------------------------------------------------------------
__global__ void sqnorm_kernel(const float* __restrict__ pts, float* __restrict__ sq,
                              int C) {
  int t = blockIdx.x * blockDim.x + threadIdx.x;
  if (t >= Bn * Nn) return;
  const float* p = pts + (long)t * C;
  float s = 0.0f;
  for (int c = 0; c < C; ++c) s += p[c] * p[c];
  sq[t] = s;
}

// ---------------------------------------------------------------------------
// kNN distance tiles via f32 WMMA 16x16x4 (pure GEMM, no selection state).
// One wave per (b, query-tile, cand-tile-in-chunk). Writes neg-squared-distance
// to dist[(b*Nn + q) * CHUNK + (cand - chunk0)].
//   f32 A/B fragment layout (ISA 7.12.2): lane l -> M/N = l&15, K = 2*(l>>4)+v.
// ---------------------------------------------------------------------------
template <int C>
__global__ void knn_dist(const float* __restrict__ pts, const float* __restrict__ sqn,
                         float* __restrict__ dist, int chunk0) {
  constexpr int CP     = (C + 3) & ~3;  // channels padded to multiple of 4
  constexpr int CTILES = CHUNK / 16;    // 32

  const int      lane = threadIdx.x & 31;
  const int      wave = threadIdx.x >> 5;
  const unsigned task = blockIdx.x * 4u + wave;
  if (task >= (unsigned)Bn * 128u * CTILES) return;   // wave-uniform

  const int b  = (int)(task >> 12);          // 128 qtiles * 32 ctiles = 4096
  const int qt = (int)((task >> 5) & 127u);
  const int ct = (int)(task & 31u);
  const int hi = lane >> 4;
  const int nl = lane & 15;

  const float* pb = pts + (long)b * Nn * C;
  const float* sb = sqn + b * Nn;

  const int q    = qt * 16 + nl;             // A-row owned by this lane
  const int cand = chunk0 + ct * 16 + nl;    // B-col owned by this lane

  v8f acc = {};
#pragma unroll
  for (int cc = 0; cc < CP; cc += 4) {
    const int c0 = cc + hi * 2;
    v2f a, bf;
    if constexpr (C % 4 == 0) {
      a  = *(const v2f*)(pb + (long)q    * C + c0);
      bf = *(const v2f*)(pb + (long)cand * C + c0);
    } else {
      a[0]  = (c0     < C) ? pb[(long)q * C + c0]        : 0.0f;
      a[1]  = (c0 + 1 < C) ? pb[(long)q * C + c0 + 1]    : 0.0f;
      bf[0] = (c0     < C) ? pb[(long)cand * C + c0]     : 0.0f;
      bf[1] = (c0 + 1 < C) ? pb[(long)cand * C + c0 + 1] : 0.0f;
    }
    acc = __builtin_amdgcn_wmma_f32_16x16x4_f32(
        false, a, false, bf, (short)0, acc, false, false);
  }

  const float sqc = sb[cand];
#pragma unroll
  for (int r = 0; r < 8; ++r) {
    const int qr = qt * 16 + r + 8 * hi;     // C/D layout: M = r + 8*hi
    const float nd = 2.0f * acc[r] - sb[qr] - sqc;
    dist[(unsigned)(b * Nn + qr) * CHUNK + (unsigned)(ct * 16 + nl)] = nd;
  }
}

// ---------------------------------------------------------------------------
// Merge one candidate chunk into each query's running top-20 (register list).
// Separate kernel -> its 40 live registers never coexist with WMMA fragments.
// ---------------------------------------------------------------------------
__global__ void topk_merge(const float* __restrict__ dist,
                           float* __restrict__ topv, int* __restrict__ topi,
                           int chunk0, int isFirst, int isLast,
                           int* __restrict__ idx) {
  const int q = blockIdx.x * blockDim.x + threadIdx.x;
  if (q >= Bn * Nn) return;

  float bv[KNN]; int bi[KNN];
  if (isFirst) {
#pragma unroll
    for (int t = 0; t < KNN; ++t) { bv[t] = -3.4e38f; bi[t] = 0; }
  } else {
#pragma unroll
    for (int t = 0; t < KNN; ++t) { bv[t] = topv[q * KNN + t]; bi[t] = topi[q * KNN + t]; }
  }

  const float* dp = dist + (size_t)q * CHUNK;
  for (int j = 0; j < CHUNK; ++j) {
    if ((j & 63) == 0 && j + 64 < CHUNK)
      __builtin_prefetch(dp + j + 64, 0, 0);   // global_prefetch_b8
    const float d = dp[j];
    if (d > bv[KNN - 1]) {
      float v = d; int id = chunk0 + j;
#pragma unroll
      for (int t = 0; t < KNN; ++t) {
        if (v > bv[t]) {
          float tv = bv[t]; int ti = bi[t];
          bv[t] = v; bi[t] = id;
          v = tv; id = ti;
        }
      }
    }
  }

  if (isLast) {
#pragma unroll
    for (int t = 0; t < KNN; ++t) idx[q * KNN + t] = bi[t];
  } else {
#pragma unroll
    for (int t = 0; t < KNN; ++t) { topv[q * KNN + t] = bv[t]; topi[q * KNN + t] = bi[t]; }
  }
}

// ---------------------------------------------------------------------------
// Edge-conv GEMM via WMMA f16->f32, 16x16x32.
//   rows  = (b,n,k) edge samples, 16 per tile  (M)
//   cols  = output channels, 16 per tile       (N)
//   K dim = 2*Cin edge-feature channels, stepped by 32 (zero padded)
// phase 0: accumulate per-channel sum / sumsq of h = f*W + bias (for BN)
// phase 1: normalize, ReLU, max-over-k folded into atomicMax on xout.
// Weights pre-converted to f16 (Wh) -> B fragment is two 16-byte loads.
// ---------------------------------------------------------------------------
template <int CIN2>
__global__ void edge_conv(const float* __restrict__ xin, const int* __restrict__ idx,
                          const _Float16* __restrict__ Wh, const float* __restrict__ bias,
                          int O, int oshift, int phase,
                          float* __restrict__ stat_sum, float* __restrict__ stat_sq,
                          const float* __restrict__ mean, const float* __restrict__ rstd,
                          const float* __restrict__ gamma, const float* __restrict__ beta,
                          float* __restrict__ xout) {
  constexpr int      CIN  = CIN2 / 2;
  constexpr int      KPAD = (CIN2 + 31) & ~31;
  constexpr unsigned ROWS_PER_B = (unsigned)Nn * KNN;        // 40960
  constexpr unsigned ROWTILES   = (unsigned)Bn * Nn * KNN / 16;

  const int      lane = threadIdx.x & 31;
  const int      wave = threadIdx.x >> 5;
  const unsigned task = blockIdx.x * 4u + wave;
  if (task >= (ROWTILES << oshift)) return;   // wave-uniform exit

  const unsigned ot      = task & ((1u << oshift) - 1u);
  const unsigned rowTile = task >> oshift;
  const int      hi      = lane >> 4;
  const int      mlo     = lane & 15;

  // Row owned by this lane for the A fragment (M = mlo)
  const unsigned rowA = rowTile * 16u + mlo;
  const unsigned bA   = rowA / ROWS_PER_B;
  const unsigned r2   = rowA % ROWS_PER_B;
  const unsigned nA   = r2 / KNN;
  const unsigned kA   = r2 % KNN;
  const int      nbr  = idx[(bA * Nn + nA) * KNN + kA];
  const float* ctr = xin + (size_t)(bA * Nn + nA) * CIN;
  const float* nbp = xin + (size_t)(bA * Nn + (unsigned)nbr) * CIN;

  const unsigned ocol = ot * 16u + mlo;   // column owned by this lane (B/C/D)

  v8f acc = {};
  for (int kk = 0; kk < KPAD; kk += 32) {
    // A fragment: 16x32 f16 (VGPR v: seg=v>>2, g=v&3; K0 = kk + seg*16 + hi*8 + 2g)
    v16h a;
#pragma unroll
    for (int v = 0; v < 8; ++v) {
      const int g  = v & 3, seg = v >> 2;
      const int k0 = kk + seg * 16 + hi * 8 + 2 * g;
      if constexpr (CIN % 2 == 0) {
        // channel pair never straddles the nb/ctr boundary (k0 even, CIN even)
        v2f pr;
        if (k0 < CIN) {
          const v2f nb2 = *(const v2f*)(nbp + k0);
          const v2f ct2 = *(const v2f*)(ctr + k0);
          pr = nb2 - ct2;
        } else {
          pr = *(const v2f*)(ctr + (k0 - CIN));
        }
        a[2 * v]     = (_Float16)pr[0];
        a[2 * v + 1] = (_Float16)pr[1];
      } else {
        auto feat = [&](int c) -> float {
          if (c >= CIN2) return 0.0f;
          return (c < CIN) ? (nbp[c] - ctr[c]) : ctr[c - CIN];
        };
        a[2 * v]     = (_Float16)feat(k0);
        a[2 * v + 1] = (_Float16)feat(k0 + 1);
      }
    }
    // B fragment: 32x16 f16; elem j<8 K=kk+hi*8+j, j>=8 K=kk+16+hi*8+(j-8)
    v16h bf;
    if constexpr (CIN2 % 32 == 0) {
      const v8h* wp = (const v8h*)(Wh + (size_t)ocol * CIN2 + kk + hi * 8);
      const v8h w0 = wp[0], w1 = wp[2];
#pragma unroll
      for (int e = 0; e < 8; ++e) { bf[e] = w0[e]; bf[8 + e] = w1[e]; }
    } else {
#pragma unroll
      for (int j = 0; j < 16; ++j) {
        const int kw = kk + (j >> 3) * 16 + hi * 8 + (j & 7);
        bf[j] = (kw < CIN2) ? Wh[(size_t)ocol * CIN2 + kw] : (_Float16)0.0f;
      }
    }
    acc = __builtin_amdgcn_wmma_f32_16x16x32_f16(
        /*neg_a=*/false, a, /*neg_b=*/false, bf,
        /*c_mod=*/(short)0, acc, /*reuse_a=*/false, /*reuse_b=*/false);
  }

  const float bo = bias[ocol];
  if (phase == 0) {
    float s = 0.0f, s2 = 0.0f;
#pragma unroll
    for (int r = 0; r < 8; ++r) {
      const float h = acc[r] + bo;
      s += h; s2 += h * h;
    }
    s  += __shfl_xor(s, 16, 32);
    s2 += __shfl_xor(s2, 16, 32);
    if (lane < 16) {
      atomicAdd(&stat_sum[ocol], s);
      atomicAdd(&stat_sq[ocol],  s2);
    }
  } else {
    const float mu = mean[ocol], rs = rstd[ocol];
    const float gm = gamma[ocol], bt = beta[ocol];
#pragma unroll
    for (int r = 0; r < 8; ++r) {
      const unsigned row = rowTile * 16u + r + 8u * hi;   // C/D layout: M = r + 8*hi
      const unsigned bb  = row / ROWS_PER_B;
      const unsigned rr  = row % ROWS_PER_B;
      const unsigned nn  = rr / KNN;
      float y = (acc[r] + bo - mu) * rs * gm + bt;
      y = fmaxf(y, 0.0f);
      // non-negative float bits are monotone as int -> safe atomicMax
      atomicMax((int*)&xout[(size_t)(bb * Nn + nn) * O + ocol], __float_as_int(y));
    }
  }
}

// ---------------------------------------------------------------------------
__global__ void finalize_stats(const float* __restrict__ s, const float* __restrict__ s2,
                               float* __restrict__ mean, float* __restrict__ rstd,
                               int O, float invCount) {
  int o = blockIdx.x * blockDim.x + threadIdx.x;
  if (o >= O) return;
  float m = s[o] * invCount;
  float v = s2[o] * invCount - m * m;
  mean[o] = m;
  rstd[o] = rsqrtf(v + 1e-5f);
}

// ---------------------------------------------------------------------------
__global__ void pool_max(const float* __restrict__ x, int C,
                         float* __restrict__ dst, int rowStride, int colOff) {
  int t = blockIdx.x * blockDim.x + threadIdx.x;
  if (t >= Bn * C) return;
  int b = t / C, c = t % C;
  float m = -3.4e38f;
  for (int n = 0; n < Nn; ++n)
    m = fmaxf(m, x[(size_t)(b * Nn + n) * C + c]);
  dst[b * rowStride + colOff + c] = m;
}

// ---------------------------------------------------------------------------
// FC + (per-channel, over-batch) BN + ReLU. Batch = 8, so BN is thread-local.
// ---------------------------------------------------------------------------
__global__ void fc_bn_relu(const float* __restrict__ in, int IN, int OUT,
                           const float* __restrict__ W, const float* __restrict__ bias,
                           const float* __restrict__ g, const float* __restrict__ bt,
                           float* __restrict__ out, int doBN) {
  int o = blockIdx.x * blockDim.x + threadIdx.x;
  if (o >= OUT) return;
  float y[Bn];
  float s = 0.0f;
  for (int b = 0; b < Bn; ++b) {
    float acc = bias[o];
    for (int c = 0; c < IN; ++c) acc += in[b * IN + c] * W[(size_t)o * IN + c];
    y[b] = acc; s += acc;
  }
  if (doBN) {
    float m = s * (1.0f / Bn), v = 0.0f;
    for (int b = 0; b < Bn; ++b) { float d = y[b] - m; v += d * d; }
    v *= (1.0f / Bn);
    float rs = rsqrtf(v + 1e-5f);
    for (int b = 0; b < Bn; ++b)
      out[b * OUT + o] = fmaxf((y[b] - m) * rs * g[o] + bt[o], 0.0f);
  } else {
    for (int b = 0; b < Bn; ++b) out[b * OUT + o] = y[b];
  }
}

// ---------------------------------------------------------------------------
// Host-side driver for one edge block
// ---------------------------------------------------------------------------
template <int CIN>
static void run_edge_block(const float* xin, float* xout, int O, int oshift,
                           const float* Wp, const float* bp,
                           const float* gp, const float* btp,
                           int* idxp, float* sqbuf, float* distb,
                           float* topv, int* topi, _Float16* Wh,
                           float* ssum, float* ssq,
                           float* mean, float* rstd, hipStream_t stream) {
  constexpr int CIN2 = 2 * CIN;

  // 1) squared norms + chunked WMMA kNN (GEMM kernel + selection kernel)
  sqnorm_kernel<<<(Bn * Nn + 255) / 256, 256, 0, stream>>>(xin, sqbuf, CIN);
  constexpr int CTILES = CHUNK / 16;
  const int distBlocks = (Bn * 128 * CTILES + 3) / 4;
  for (int chunk0 = 0; chunk0 < Nn; chunk0 += CHUNK) {
    knn_dist<CIN><<<distBlocks, 128, 0, stream>>>(xin, sqbuf, distb, chunk0);
    topk_merge<<<(Bn * Nn + 255) / 256, 256, 0, stream>>>(
        distb, topv, topi, chunk0,
        /*isFirst=*/chunk0 == 0, /*isLast=*/chunk0 + CHUNK >= Nn, idxp);
  }

  // 2) f16 weights + zero BN accumulators
  cvt_w_f16<<<(O * CIN2 + 255) / 256, 256, 0, stream>>>(Wp, Wh, O * CIN2);
  zero_f32<<<2, 256, 0, stream>>>(ssum, 256);
  zero_f32<<<2, 256, 0, stream>>>(ssq, 256);

  const unsigned rowTiles = (unsigned)Bn * Nn * KNN / 16;
  const unsigned tasks    = rowTiles << oshift;
  const int      blocks   = (int)((tasks + 3) / 4);

  // 3) pass 0: WMMA GEMM + per-channel sum/sumsq
  edge_conv<CIN2><<<blocks, 128, 0, stream>>>(xin, idxp, Wh, bp, O, oshift, 0,
                                              ssum, ssq, mean, rstd, gp, btp, xout);

  // 4) mean / rstd
  finalize_stats<<<(O + 63) / 64, 64, 0, stream>>>(
      ssum, ssq, mean, rstd, O, 1.0f / ((float)Bn * Nn * KNN));

  // 5) zero output (post-ReLU max is >= 0, so 0-init is a valid identity)
  zero_f32<<<(Bn * Nn * O + 255) / 256, 256, 0, stream>>>(xout, Bn * Nn * O);

  // 6) pass 1: WMMA GEMM + BN + ReLU + max-over-k
  edge_conv<CIN2><<<blocks, 128, 0, stream>>>(xin, idxp, Wh, bp, O, oshift, 1,
                                              ssum, ssq, mean, rstd, gp, btp, xout);
}

// ---------------------------------------------------------------------------
// Entry point
// ---------------------------------------------------------------------------
extern "C" void kernel_launch(void* const* d_in, const int* in_sizes, int n_in,
                              void* d_out, int out_size, void* d_ws, size_t ws_size,
                              hipStream_t stream) {
  (void)in_sizes; (void)n_in; (void)out_size; (void)ws_size;

  const float* x       = (const float*)d_in[0];   // (B,1,N,3) contiguous == (B,N,3)
  const float* conv1_w = (const float*)d_in[1];
  const float* conv1_b = (const float*)d_in[2];
  const float* bn1_g   = (const float*)d_in[3];
  const float* bn1_b   = (const float*)d_in[4];
  const float* conv2_w = (const float*)d_in[5];
  const float* conv2_b = (const float*)d_in[6];
  const float* bn2_g   = (const float*)d_in[7];
  const float* bn2_b   = (const float*)d_in[8];
  const float* conv3_w = (const float*)d_in[9];
  const float* conv3_b = (const float*)d_in[10];
  const float* bn3_g   = (const float*)d_in[11];
  const float* bn3_b   = (const float*)d_in[12];
  const float* conv4_w = (const float*)d_in[13];
  const float* conv4_b = (const float*)d_in[14];
  const float* bn4_g   = (const float*)d_in[15];
  const float* bn4_b   = (const float*)d_in[16];
  const float* fc1_w   = (const float*)d_in[17];
  const float* fc1_b   = (const float*)d_in[18];
  const float* bnf1_g  = (const float*)d_in[19];
  const float* bnf1_b  = (const float*)d_in[20];
  const float* fc2_w   = (const float*)d_in[21];
  const float* fc2_b   = (const float*)d_in[22];
  const float* bnf2_g  = (const float*)d_in[23];
  const float* bnf2_b  = (const float*)d_in[24];
  const float* fc3_w   = (const float*)d_in[25];
  const float* fc3_b   = (const float*)d_in[26];
  const float* bnf3_g  = (const float*)d_in[27];
  const float* bnf3_b  = (const float*)d_in[28];
  const float* fc4_w   = (const float*)d_in[29];
  const float* fc4_b   = (const float*)d_in[30];

  // ---- carve workspace (~72 MB) ----
  char*  ws  = (char*)d_ws;
  size_t off = 0;
  auto alloc = [&](size_t bytes) -> void* {
    void* p = ws + off;
    off = (off + bytes + 255) & ~(size_t)255;
    return p;
  };
  int*      idxb  = (int*)     alloc((size_t)Bn * Nn * KNN * sizeof(int));
  float*    x1    = (float*)   alloc((size_t)Bn * Nn * 64  * sizeof(float));
  float*    x2    = (float*)   alloc((size_t)Bn * Nn * 64  * sizeof(float));
  float*    x3    = (float*)   alloc((size_t)Bn * Nn * 128 * sizeof(float));
  float*    x4    = (float*)   alloc((size_t)Bn * Nn * 256 * sizeof(float));
  float*    sqb   = (float*)   alloc((size_t)Bn * Nn * sizeof(float));
  float*    distb = (float*)   alloc((size_t)Bn * Nn * CHUNK * sizeof(float));
  float*    topv  = (float*)   alloc((size_t)Bn * Nn * KNN * sizeof(float));
  int*      topi  = (int*)     alloc((size_t)Bn * Nn * KNN * sizeof(int));
  _Float16* Wh    = (_Float16*)alloc((size_t)256 * 256 * sizeof(_Float16));
  float*    ssum  = (float*)   alloc(256 * sizeof(float));
  float*    ssq   = (float*)   alloc(256 * sizeof(float));
  float*    meanb = (float*)   alloc(256 * sizeof(float));
  float*    rstdb = (float*)   alloc(256 * sizeof(float));
  float*    hcat  = (float*)   alloc((size_t)Bn * 448 * sizeof(float));
  float*    f1    = (float*)   alloc((size_t)Bn * 256 * sizeof(float));
  float*    f2    = (float*)   alloc((size_t)Bn * 128 * sizeof(float));
  float*    f3    = (float*)   alloc((size_t)Bn * 16  * sizeof(float));

  // ---- four EdgeConv blocks ----
  run_edge_block<3>  (x,  x1,  64, 2, conv1_w, conv1_b, bn1_g, bn1_b,
                      idxb, sqb, distb, topv, topi, Wh, ssum, ssq, meanb, rstdb, stream);
  run_edge_block<64> (x1, x2,  64, 2, conv2_w, conv2_b, bn2_g, bn2_b,
                      idxb, sqb, distb, topv, topi, Wh, ssum, ssq, meanb, rstdb, stream);
  run_edge_block<64> (x2, x3, 128, 3, conv3_w, conv3_b, bn3_g, bn3_b,
                      idxb, sqb, distb, topv, topi, Wh, ssum, ssq, meanb, rstdb, stream);
  run_edge_block<128>(x3, x4, 256, 4, conv4_w, conv4_b, bn4_g, bn4_b,
                      idxb, sqb, distb, topv, topi, Wh, ssum, ssq, meanb, rstdb, stream);

  // ---- global max pools -> concat [x256 | x128 | x64] (row stride 448) ----
  pool_max<<<(Bn * 256 + 127) / 128, 128, 0, stream>>>(x4, 256, hcat, 448, 0);
  pool_max<<<(Bn * 128 + 127) / 128, 128, 0, stream>>>(x3, 128, hcat, 448, 256);
  pool_max<<<(Bn *  64 + 127) / 128, 128, 0, stream>>>(x2,  64, hcat, 448, 384);

  // ---- FC head ----
  fc_bn_relu<<<1, 256, 0, stream>>>(hcat, 448, 256, fc1_w, fc1_b, bnf1_g, bnf1_b, f1, 1);
  fc_bn_relu<<<1, 128, 0, stream>>>(f1,   256, 128, fc2_w, fc2_b, bnf2_g, bnf2_b, f2, 1);
  fc_bn_relu<<<1,  32, 0, stream>>>(f2,   128,  16, fc3_w, fc3_b, bnf3_g, bnf3_b, f3, 1);
  fc_bn_relu<<<1,  32, 0, stream>>>(f3,    16,   1, fc4_w, fc4_b, nullptr, nullptr,
                                    (float*)d_out, 0);
}
